// GINModel_45320494907956
// MI455X (gfx1250) — matrix operand order repaired and checked by reading, not compiled, not run on hardware
//
#include <hip/hip_runtime.h>
#include <hip/hip_bf16.h>

#define H 128
#define LAYERS 3
#define TCLS 10
#define BN_EPS 1e-5f

typedef __attribute__((ext_vector_type(16))) __bf16 v16bf;
typedef __attribute__((ext_vector_type(8)))  float  v8f;

union BF16x16 { v16bf v; uint4 u[2]; };

// ---------------- Param preparation ----------------

// Wt[s][n][k] = (bf16) W_sub[layer][k][n]   (s = layer*2 + sub)
__global__ void prep_weights(const float* __restrict__ W1,
                             const float* __restrict__ W2,
                             __bf16* __restrict__ Wt) {
    int t = blockIdx.x * blockDim.x + threadIdx.x;
    if (t >= 6 * H * H) return;
    int s = t >> 14;          // /16384
    int r = t & 16383;
    int n = r >> 7;
    int k = r & 127;
    int layer = s >> 1;
    const float* W = (s & 1) ? W2 : W1;
    Wt[t] = (__bf16)W[layer * H * H + k * H + n];
}

// Fold BN (and outer BN into sub==1 for layers 0..L-2) into scale/shift.
__global__ void prep_affine(const float* b1, const float* g1, const float* be1,
                            const float* m1, const float* v1,
                            const float* b2, const float* g2, const float* be2,
                            const float* m2, const float* v2,
                            const float* og, const float* ob,
                            const float* om, const float* ov,
                            float* __restrict__ bias, float* __restrict__ scale,
                            float* __restrict__ shift) {
    int t = blockIdx.x * blockDim.x + threadIdx.x;
    if (t >= 6 * H) return;
    int s = t >> 7;
    int n = t & 127;
    int i = s >> 1;
    int sub = s & 1;
    const float *b, *g, *be, *m, *v;
    if (sub == 0) { b = b1; g = g1; be = be1; m = m1; v = v1; }
    else          { b = b2; g = g2; be = be2; m = m2; v = v2; }
    int idx = i * H + n;
    float a = g[idx] * rsqrtf(v[idx] + BN_EPS);
    float c = be[idx] - m[idx] * a;
    if (sub == 1 && i < LAYERS - 1) {
        float ao = og[idx] * rsqrtf(ov[idx] + BN_EPS);
        float co = ob[idx] - om[idx] * ao;
        a *= ao;
        c = c * ao + co;
    }
    bias[t] = b[idx];
    scale[t] = a;
    shift[t] = c;
}

// ---------------- Data movement kernels ----------------

__global__ void copy_f4(const float4* __restrict__ src, float4* __restrict__ dst,
                        long long n) {
    long long i = (long long)blockIdx.x * blockDim.x + threadIdx.x;
    if (i < n) dst[i] = src[i];
}

__global__ void zero_f(float* __restrict__ p, int n) {
    int i = blockIdx.x * blockDim.x + threadIdx.x;
    if (i < n) p[i] = 0.0f;
}

// One wave per edge; lane handles 4 channels. hsum pre-initialized with h.
__global__ void edge_scatter(const float* __restrict__ h, float* hsum,
                             const int* __restrict__ src,
                             const int* __restrict__ dst, int E) {
    long long tid = (long long)blockIdx.x * blockDim.x + threadIdx.x;
    int e = (int)(tid >> 5);
    if (e >= E) return;
    int c = ((int)tid & 31) << 2;
    int s = src[e], d = dst[e];
    float4 val = *(const float4*)(h + (size_t)s * H + c);
    float* p = hsum + (size_t)d * H + c;
    atomicAdd(p + 0, val.x);
    atomicAdd(p + 1, val.y);
    atomicAdd(p + 2, val.z);
    atomicAdd(p + 3, val.w);
}

// One wave per node; lane handles 4 channels of the pooled segment sum.
__global__ void pool_scatter(const float* __restrict__ h, float* pooled,
                             const int* __restrict__ batch, int N) {
    long long tid = (long long)blockIdx.x * blockDim.x + threadIdx.x;
    int node = (int)(tid >> 5);
    if (node >= N) return;
    int c = ((int)tid & 31) << 2;
    int g = batch[node];
    float4 val = *(const float4*)(h + (size_t)node * H + c);
    float* p = pooled + (size_t)g * H + c;
    atomicAdd(p + 0, val.x);
    atomicAdd(p + 1, val.y);
    atomicAdd(p + 2, val.z);
    atomicAdd(p + 3, val.w);
}

// out[g][t] = b_out[t] + sum_k pooled[g][k] * W_out[k][t]
__global__ void final_gemm(const float* __restrict__ pooled,
                           const float* __restrict__ Wout,
                           const float* __restrict__ bout,
                           float* __restrict__ out, int G) {
    int tid = blockIdx.x * blockDim.x + threadIdx.x;
    if (tid >= G * TCLS) return;
    int g = tid / TCLS, t = tid % TCLS;
    float s = bout[t];
    const float* pr = pooled + (size_t)g * H;
    #pragma unroll 8
    for (int k = 0; k < H; ++k) s += pr[k] * Wout[k * TCLS + t];
    out[tid] = s;
}

// ---------------- Fused WMMA MLP: out = relu(in @ W + bias) * scale + shift ----
// Wt is [n][k] (pre-transposed, bf16). 8 waves/block, each wave: 16 rows x 128 cols.
__global__ void __launch_bounds__(256)
gin_mlp_wmma(const float* __restrict__ in, float* __restrict__ out,
             const __bf16* __restrict__ Wt,
             const float* __restrict__ bias, const float* __restrict__ scale,
             const float* __restrict__ shift, int nrows) {
    __shared__ uint4 sW4[2048];               // 32 KB: full 128x128 bf16 weight
    {
        const uint4* g = (const uint4*)Wt;
        for (int j = threadIdx.x; j < 2048; j += 256) sW4[j] = g[j];
    }
    __syncthreads();
    const __bf16* sW = (const __bf16*)sW4;

    int wave = threadIdx.x >> 5;
    int lane = threadIdx.x & 31;
    int tile = blockIdx.x * 8 + wave;
    if (tile * 16 >= nrows) return;           // wave-uniform exit: EXEC stays all-1
    int m0 = tile * 16;
    int half = lane >> 4;
    int l16 = lane & 15;
    int mrow = m0 + l16;
    if (mrow >= nrows) mrow = nrows - 1;      // clamp loads; stores guarded below
    const float* arow = in + (size_t)mrow * H;

    v8f acc[8] = {};

    #pragma unroll
    for (int kc = 0; kc < H; kc += 32) {
        // A fragment: ISA layout = two contiguous 8-elem runs per lane
        BF16x16 a;
        {
            float fs[16];
            const float4* p0 = (const float4*)(arow + kc + half * 8);
            const float4* p1 = (const float4*)(arow + kc + 16 + half * 8);
            *(float4*)&fs[0]  = p0[0];
            *(float4*)&fs[4]  = p0[1];
            *(float4*)&fs[8]  = p1[0];
            *(float4*)&fs[12] = p1[1];
            #pragma unroll
            for (int j = 0; j < 16; ++j) a.v[j] = (__bf16)fs[j];
        }
        #pragma unroll
        for (int t = 0; t < 8; ++t) {
            // B fragment: ISA layout = two contiguous 8-elem runs per lane (K = half*8.., 16+half*8..)
            int n = t * 16 + l16;
            int o0 = n * H + kc + half * 8;   // bf16 element offset, 16B aligned
            BF16x16 b;
            b.u[0] = *(const uint4*)&sW[o0];
            b.u[1] = *(const uint4*)&sW[o0 + 16];
            acc[t] = __builtin_amdgcn_wmma_f32_16x16x32_bf16(
                false, a.v, false, b.v, (short)0, acc[t], false, false);
        }
    }

    // Epilogue: C/D layout -> row = m0 + half*8 + r, col = t*16 + l16
    #pragma unroll
    for (int t = 0; t < 8; ++t) {
        int n = t * 16 + l16;
        float bsv = bias[n], scv = scale[n], shv = shift[n];
        #pragma unroll
        for (int r = 0; r < 8; ++r) {
            int row = m0 + half * 8 + r;
            if (row < nrows) {
                float z = acc[t][r] + bsv;
                z = z > 0.0f ? z : 0.0f;
                out[(size_t)row * H + n] = z * scv + shv;
            }
        }
    }
}

// ---------------- Launch ----------------

extern "C" void kernel_launch(void* const* d_in, const int* in_sizes, int n_in,
                              void* d_out, int out_size, void* d_ws, size_t ws_size,
                              hipStream_t stream) {
    const float* x      = (const float*)d_in[0];
    const int*   edge   = (const int*)d_in[1];
    const int*   batch  = (const int*)d_in[2];
    const float* W1     = (const float*)d_in[3];
    const float* b1     = (const float*)d_in[4];
    const float* g1     = (const float*)d_in[5];
    const float* be1    = (const float*)d_in[6];
    const float* m1     = (const float*)d_in[7];
    const float* v1     = (const float*)d_in[8];
    const float* W2     = (const float*)d_in[9];
    const float* b2     = (const float*)d_in[10];
    const float* g2     = (const float*)d_in[11];
    const float* be2    = (const float*)d_in[12];
    const float* m2     = (const float*)d_in[13];
    const float* v2     = (const float*)d_in[14];
    const float* og     = (const float*)d_in[15];
    const float* ob     = (const float*)d_in[16];
    const float* om     = (const float*)d_in[17];
    const float* ov     = (const float*)d_in[18];
    const float* Wout   = (const float*)d_in[19];
    const float* bout   = (const float*)d_in[20];

    const int N = in_sizes[0] / H;
    const int E = in_sizes[1] / 2;
    const int G = out_size / TCLS;
    const int* src = edge;
    const int* dst = edge + E;

    // workspace layout
    char* ws = (char*)d_ws;
    size_t nodeBytes = (size_t)N * H * sizeof(float);
    float* bufA = (float*)ws;
    float* bufB = (float*)(ws + nodeBytes);
    float* bufC = (float*)(ws + 2 * nodeBytes);
    float* pooled = (float*)(ws + 3 * nodeBytes);
    char* pp = ws + 3 * nodeBytes + (size_t)G * H * sizeof(float);
    __bf16* Wt   = (__bf16*)pp;                       // 6 * 128*128 bf16
    float* biasF = (float*)(pp + (size_t)6 * H * H * 2);
    float* scaleF = biasF + 6 * H;
    float* shiftF = scaleF + 6 * H;

    // 1. fold params
    prep_weights<<<(6 * H * H + 255) / 256, 256, 0, stream>>>(W1, W2, Wt);
    prep_affine<<<3, 256, 0, stream>>>(b1, g1, be1, m1, v1,
                                       b2, g2, be2, m2, v2,
                                       og, ob, om, ov, biasF, scaleF, shiftF);

    const int mtiles = (N + 15) / 16;
    const int gblocks = (mtiles + 7) / 8;
    const long long cthreads = (long long)N * 32;
    const long long ethreads = (long long)E * 32;

    const float* hin = x;
    for (int i = 0; i < LAYERS; ++i) {
        // hsum = h + scatter_sum(h[src] -> dst)
        copy_f4<<<(int)((cthreads + 255) / 256), 256, 0, stream>>>(
            (const float4*)hin, (float4*)bufA, cthreads);
        edge_scatter<<<(int)((ethreads + 255) / 256), 256, 0, stream>>>(
            hin, bufA, src, dst, E);
        // MLP sub-layer 1 and 2 (BN folded, outer BN composed into sub 2)
        gin_mlp_wmma<<<gblocks, 256, 0, stream>>>(
            bufA, bufB, Wt + (size_t)(2 * i) * H * H,
            biasF + (2 * i) * H, scaleF + (2 * i) * H, shiftF + (2 * i) * H, N);
        gin_mlp_wmma<<<gblocks, 256, 0, stream>>>(
            bufB, bufC, Wt + (size_t)(2 * i + 1) * H * H,
            biasF + (2 * i + 1) * H, scaleF + (2 * i + 1) * H, shiftF + (2 * i + 1) * H, N);
        hin = bufC;
    }

    // 2. pooling + output head
    zero_f<<<(G * H + 255) / 256, 256, 0, stream>>>(pooled, G * H);
    pool_scatter<<<(int)((cthreads + 255) / 256), 256, 0, stream>>>(
        hin, pooled, batch, N);
    final_gemm<<<(G * TCLS + 255) / 256, 256, 0, stream>>>(
        pooled, Wout, bout, (float*)d_out, G);
}
